// GCN_Unit_35081292873882
// MI455X (gfx1250) — matrix operand order repaired and checked by reading, not compiled, not run on hardware
//
#include <hip/hip_runtime.h>
#include <hip/hip_bf16.h>

typedef __attribute__((ext_vector_type(16))) _Float16 v16h;
typedef __attribute__((ext_vector_type(8)))  _Float16 v8h;
typedef __attribute__((ext_vector_type(8)))  float    v8f;

#define BATCH 2
#define CIN   64
#define CMID  64
#define HIN   96
#define HOUT  48
#define NNODE 2304            // 48*48
#define BN    (BATCH*NNODE)   // 4608
#define NNBR  7
#define ALPHA 0.2f

// ---- gfx1250 WMMA f16 wave32 fragments as loaded here (A/B: contiguous k = 16h+s, same in both) ----
// A 16x32: lane&15 = M; one contiguous 16-half run at row + k0 + (lane&16?16:0).
// B 32x16: lane&15 = N; one contiguous 16-half run at col + k0 + (lane&16?16:0).
// D f32:   lane&15 = N; element r -> M = r + (lane&16?8:0).
__device__ __forceinline__ int d_m(int lane, int r) { return r + ((lane & 16) ? 8 : 0); }

__device__ __forceinline__ v16h load_a_frag(const _Float16* row_k0, int lane) {
    return *(const v16h*)(row_k0 + ((lane & 16) ? 16 : 0));
}
__device__ __forceinline__ v16h load_b_frag(const _Float16* col_k0, int lane) {
    return *(const v16h*)(col_k0 + ((lane & 16) ? 16 : 0));
}
__device__ __forceinline__ v8f wmma_f16(v16h a, v16h b, v8f c) {
    return __builtin_amdgcn_wmma_f32_16x16x32_f16(false, a, false, b, (short)0, c, false, false);
}

// ================= Kernel 1: stride-2 3x3 conv, implicit im2col, WMMA =================
// Block = 128 thr (4 waves = 4 oc tiles) per 16-node tile. Halo tile async-DMA'd to LDS
// (f32), converted once to f16 channel-innermost so A-frags are contiguous ds_load_b128s.
// Weights staged transposed as f16. K=576 fully unrolled -> 18 WMMAs, zero converts in loop.
__global__ __launch_bounds__(128)
void k_conv_s2_wmma(const float* __restrict__ x, const float* __restrict__ cw,
                    const float* __restrict__ cb,
                    float* __restrict__ feat, _Float16* __restrict__ featH) {
    extern __shared__ char smem[];
    float*    xs32 = (float*)smem;                          // [3][33][64] f32 halo (async target)
    _Float16* xs16 = (_Float16*)(smem + 25344);             // [3][33][64] f16
    _Float16* wshT = (_Float16*)(smem + 25344 + 12672);     // [64 oc][576 k]

    const int tid  = threadIdx.x;
    const int lane = tid & 31;
    const int wave = tid >> 5;
    const int b    = blockIdx.x / 144;
    const int m0   = (blockIdx.x % 144) * 16;
    const int oy   = m0 / HOUT;          // 48%16==0: all 16 nodes share oy
    const int ox0  = m0 % HOUT;

    for (int idx = tid; idx < 3 * 33 * 64; idx += 128) xs32[idx] = 0.f;
    __syncthreads();

    // async DMA in-bounds halo elements into LDS (ASYNCcnt path); lanes walk xx for coalescing
    for (int idx = tid; idx < 64 * 3 * 33; idx += 128) {
        int xx = idx % 33; int t2 = idx / 33;
        int r  = t2 % 3;   int ic = t2 / 3;
        int iy = 2 * oy - 1 + r;
        int ix = 2 * ox0 - 1 + xx;
        if (iy >= 0 && iy < HIN && ix >= 0 && ix < HIN) {
            unsigned ldsoff = (unsigned)(uintptr_t)&xs32[(r * 33 + xx) * 64 + ic];
            const float* gp = &x[((size_t)(b * CIN + ic) * HIN + iy) * HIN + ix];
            asm volatile("global_load_async_to_lds_b32 %0, %1, off"
                         :: "v"(ldsoff), "v"(gp) : "memory");
        }
    }
    // stage weights transposed: wshT[oc][k], k = tap*64 + ic
    for (int idx = tid; idx < 64 * 576; idx += 128) {
        int k = idx % 576; int oc = idx / 576;
        int tap = k >> 6, ic = k & 63;
        wshT[idx] = (_Float16)cw[(size_t)(oc * 64 + ic) * 9 + tap];
    }
    asm volatile("s_wait_asynccnt 0x0" ::: "memory");
    __syncthreads();
    // one-shot f32 -> f16 convert (paired)
    for (int p = tid; p < 3 * 33 * 32; p += 128) {
        int e = 2 * p;
        xs16[e]     = (_Float16)xs32[e];
        xs16[e + 1] = (_Float16)xs32[e + 1];
    }
    __syncthreads();

    const int n    = wave * 16 + (lane & 15);
    const int mloc = lane & 15;
    const _Float16* wrow = wshT + (size_t)n * 576;

    v8f acc = {0.f,0.f,0.f,0.f,0.f,0.f,0.f,0.f};
#pragma unroll
    for (int kc = 0; kc < 576; kc += 32) {
        const int tap = kc >> 6, ic0 = kc & 63;   // tap constant within a 32-chunk
        const int ky = tap / 3, kx = tap % 3;
        v16h af = load_a_frag(xs16 + ((ky * 33) + (2 * mloc + kx)) * 64 + ic0, lane);
        v16h bf = load_b_frag(wrow + kc, lane);
        acc = wmma_f16(af, bf, acc);
    }
#pragma unroll
    for (int r = 0; r < 8; r++) {
        int mm = m0 + d_m(lane, r);
        float v = acc[r] + cb[n];
        size_t o = (size_t)(b * NNODE + mm) * CMID + n;
        feat[o]  = v;
        featH[o] = (_Float16)v;
    }
}

// ================= Kernel 2: per-node squared norm (float4 loads) =================
__global__ void k_rowsq(const float* __restrict__ feat, float* __restrict__ sq) {
    int i = blockIdx.x * blockDim.x + threadIdx.x;
    if (i >= BN) return;
    const float4* f4 = (const float4*)(feat + (size_t)i * CMID);
    float s = 0.f;
#pragma unroll
    for (int c = 0; c < 16; c++) {
        float4 v = f4[c];
        s += v.x * v.x + v.y * v.y + v.z * v.z + v.w * v.w;
    }
    sq[i] = s;
}

// ============ Kernel 3: WMMA Gram stripe in 147KB LDS + stable top-7 selection ============
__global__ __launch_bounds__(256)
void k_knn7_wmma(const _Float16* __restrict__ featH, const float* __restrict__ sq,
                 int* __restrict__ nbr) {
    extern __shared__ float drow[];                 // [16][NNODE]
    const int b    = blockIdx.x / 144;
    const int i0   = (blockIdx.x % 144) * 16;
    const int wave = threadIdx.x >> 5;
    const int lane = threadIdx.x & 31;
    const _Float16* fb = featH + (size_t)b * NNODE * CMID;

    const _Float16* arow = fb + (size_t)(i0 + (lane & 15)) * CMID;
    v16h a0 = load_a_frag(arow, lane);
    v16h a1 = load_a_frag(arow + 32, lane);

    for (int jt = wave; jt < 144; jt += 8) {
        int j0 = jt * 16;
        int nc = j0 + (lane & 15);
        const _Float16* brow = fb + (size_t)nc * CMID;
        v16h b0 = load_b_frag(brow, lane);
        v16h b1 = load_b_frag(brow + 32, lane);
        v8f acc = {0.f,0.f,0.f,0.f,0.f,0.f,0.f,0.f};
        acc = wmma_f16(a0, b0, acc);
        acc = wmma_f16(a1, b1, acc);
        float sqj = sq[b * NNODE + nc];
#pragma unroll
        for (int r = 0; r < 8; r++) {
            int mm = d_m(lane, r);
            float d2 = sq[b * NNODE + i0 + mm] + sqj - 2.f * acc[r];
            drow[mm * NNODE + nc] = sqrtf(fmaxf(d2, 0.f));
        }
    }
    __syncthreads();

    // stable 7-smallest per row (value,index lexicographic == double-argsort ranks<=6)
    for (int rr = 0; rr < 2; rr++) {
        int row = wave * 2 + rr;
        float* dr = drow + (size_t)row * NNODE;
        int gi = b * NNODE + i0 + row;
        for (int s = 0; s < NNBR; s++) {
            float best = 3.4e38f; int bj = 0x7fffffff;
            for (int j = lane; j < NNODE; j += 32) {
                float v = dr[j];
                if (v < best || (v == best && j < bj)) { best = v; bj = j; }
            }
#pragma unroll
            for (int off = 16; off > 0; off >>= 1) {
                float ov = __shfl_xor(best, off, 32);
                int   oj = __shfl_xor(bj,   off, 32);
                if (ov < best || (ov == best && oj < bj)) { best = ov; bj = oj; }
            }
            if (lane == 0) nbr[(size_t)gi * NNBR + s] = bj;
            dr[bj] = 3.4e38f;
        }
    }
}

// ====== Kernel 4: node-feature GEMM Wh = Hin @ W, K templated (full unroll), 4 waves ======
template <int K>
__global__ __launch_bounds__(128)
void k_gemm_wmma(const _Float16* __restrict__ hinH, const float* __restrict__ W,
                 float* __restrict__ out) {
    __shared__ __align__(32) _Float16 wht[64 * K];   // W transposed [n][k] as f16
    const int tid  = threadIdx.x;
    const int lane = tid & 31;
    const int wave = tid >> 5;
    for (int idx = tid; idx < 64 * K; idx += 128) {
        int n = idx / K, k = idx % K;
        wht[idx] = (_Float16)W[(size_t)k * CMID + n];
    }
    __syncthreads();

    const int m0 = blockIdx.x * 16;
    const int n  = wave * 16 + (lane & 15);
    const _Float16* arow = hinH + (size_t)(m0 + (lane & 15)) * K;
    const _Float16* brow = wht + (size_t)n * K;

    v8f acc = {0.f,0.f,0.f,0.f,0.f,0.f,0.f,0.f};
#pragma unroll
    for (int k0 = 0; k0 < K; k0 += 32) {
        v16h af = load_a_frag(arow + k0, lane);
        v16h bf = load_b_frag(brow + k0, lane);
        acc = wmma_f16(af, bf, acc);
    }
#pragma unroll
    for (int r = 0; r < 8; r++)
        out[(size_t)(m0 + d_m(lane, r)) * CMID + n] = acc[r];
}

// ====== Kernel 5: attention scores s1 = Wh·a[:64], s2 = Wh·a[64:] (float4) ======
__global__ void k_attn_scores(const float* __restrict__ Wh, const float* __restrict__ a,
                              float* __restrict__ s1, float* __restrict__ s2) {
    int i = blockIdx.x * blockDim.x + threadIdx.x;
    if (i >= BN) return;
    const float4* w4 = (const float4*)(Wh + (size_t)i * CMID);
    const float4* a4 = (const float4*)a;
    float x1 = 0.f, x2 = 0.f;
#pragma unroll
    for (int c = 0; c < 16; c++) {
        float4 v = w4[c], p = a4[c], q = a4[16 + c];
        x1 += v.x * p.x + v.y * p.y + v.z * p.z + v.w * p.w;
        x2 += v.x * q.x + v.y * q.y + v.z * q.z + v.w * q.w;
    }
    s1[i] = x1; s2[i] = x2;
}

// ====== Kernel 6: 7-sparse attention softmax + aggregate (+ELU) ======
// chw==0: write f32 (+f16 copy) to [B*N][ostride] at column ooff.
// chw==1: write f16 [B][C][N] (input layout for the WMMA transpose conv).
__global__ __launch_bounds__(256)
void k_gat_aggregate(const float* __restrict__ Wh, const float* __restrict__ s1,
                     const float* __restrict__ s2, const int* __restrict__ nbr,
                     float* __restrict__ out, _Float16* __restrict__ outH,
                     int ostride, int ooff, int do_elu, int chw) {
    int t = blockIdx.x * blockDim.x + threadIdx.x;
    int i = t >> 6, c = t & 63;
    if (i >= BN) return;
    int b = i / NNODE;
    float si = s1[i];
    float e[NNBR]; int jn[NNBR];
    float mx = -3.4e38f;
#pragma unroll
    for (int s = 0; s < NNBR; s++) {
        int j = b * NNODE + nbr[(size_t)i * NNBR + s];
        jn[s] = j;
        float v = si + s2[j];
        v = (v > 0.f) ? v : ALPHA * v;          // leaky_relu
        e[s] = v; mx = fmaxf(mx, v);
    }
    float den = 0.f;
#pragma unroll
    for (int s = 0; s < NNBR; s++) { e[s] = __expf(e[s] - mx); den += e[s]; }
    float acc = 0.f;
#pragma unroll
    for (int s = 0; s < NNBR; s++) acc += e[s] * Wh[(size_t)jn[s] * CMID + c];
    acc /= den;
    if (do_elu) acc = (acc > 0.f) ? acc : (__expf(acc) - 1.f);
    if (chw) {
        outH[((size_t)(b * CMID + c)) * NNODE + (i - b * NNODE)] = (_Float16)acc;
    } else {
        size_t o = (size_t)i * ostride + ooff + c;
        out[o] = acc;
        if (outH) outH[o] = (_Float16)acc;
    }
}

// ====== Kernel 7: transpose conv (k=3,s=2,p=1,op=1) as stride-1 conv over the ======
// zero-dilated 98x98 grid with flipped/transposed weights -- same implicit-GEMM WMMA
// structure as the forward conv. Block = 16 output px (one row) x 64 oc, K=576 unrolled.
__global__ __launch_bounds__(128)
void k_tconv_wmma(const _Float16* __restrict__ gH, const float* __restrict__ tw,
                  const float* __restrict__ tb, float* __restrict__ out) {
    extern __shared__ char smem[];
    _Float16* us   = (_Float16*)smem;                 // [3][18][64] dilated/padded tile, f16
    _Float16* wshT = (_Float16*)(smem + 3 * 18 * 64 * 2); // [64 oc][576 k] flipped taps

    const int tid  = threadIdx.x;
    const int lane = tid & 31;
    const int wave = tid >> 5;
    const int b    = blockIdx.x / (96 * 6);
    const int rem  = blockIdx.x % (96 * 6);
    const int oy   = rem / 6;
    const int ox0  = (rem % 6) * 16;

    // dilated/padded input: U[p] nonzero iff p odd && p <= 95; U[p] = g[(p-1)/2]
    for (int idx = tid; idx < 3 * 18 * 64; idx += 128) {
        int ic = idx & 63; int t2 = idx >> 6;
        int xx = t2 % 18;  int r  = t2 / 18;
        int py = oy + r, px = ox0 + xx;
        _Float16 v = (_Float16)0.f;
        if ((py & 1) && py <= 95 && (px & 1) && px <= 95) {
            int iy = (py - 1) >> 1, ix = (px - 1) >> 1;
            v = gH[((size_t)(b * CMID + ic)) * NNODE + iy * HOUT + ix];
        }
        us[idx] = v;
    }
    // wshT[oc][tap*64+ic] = tw[ic][oc][2-ky][2-kx]   (torch [in,out,kh,kw], flipped)
    for (int idx = tid; idx < 64 * 576; idx += 128) {
        int k = idx % 576, oc = idx / 576;
        int tap = k >> 6, ic = k & 63;
        int ky = tap / 3, kx = tap % 3;
        wshT[idx] = (_Float16)tw[((size_t)(ic * 64 + oc) * 3 + (2 - ky)) * 3 + (2 - kx)];
    }
    __syncthreads();

    const int n    = wave * 16 + (lane & 15);
    const int mloc = lane & 15;
    const _Float16* wrow = wshT + (size_t)n * 576;

    v8f acc = {0.f,0.f,0.f,0.f,0.f,0.f,0.f,0.f};
#pragma unroll
    for (int kc = 0; kc < 576; kc += 32) {
        const int tap = kc >> 6, ic0 = kc & 63;
        const int ky = tap / 3, kx = tap % 3;
        v16h af = load_a_frag(us + ((ky * 18) + (mloc + kx)) * 64 + ic0, lane);
        v16h bf = load_b_frag(wrow + kc, lane);
        acc = wmma_f16(af, bf, acc);
    }
#pragma unroll
    for (int r = 0; r < 8; r++) {
        int ox = ox0 + d_m(lane, r);
        out[(((size_t)(b * CMID + n)) * HIN + oy) * HIN + ox] = acc[r] + tb[n];
    }
}

extern "C" void kernel_launch(void* const* d_in, const int* in_sizes, int n_in,
                              void* d_out, int out_size, void* d_ws, size_t ws_size,
                              hipStream_t stream) {
    const float* x      = (const float*)d_in[0];
    const float* conv_w = (const float*)d_in[1];
    const float* conv_b = (const float*)d_in[2];
    const float* W1     = (const float*)d_in[3];
    const float* a1     = (const float*)d_in[4];
    const float* W2     = (const float*)d_in[5];
    const float* a2     = (const float*)d_in[6];
    const float* Wout   = (const float*)d_in[7];
    const float* aout   = (const float*)d_in[8];
    const float* tw     = (const float*)d_in[9];
    const float* tb     = (const float*)d_in[10];

    char* ws = (char*)d_ws;
    size_t off = 0;
    auto alloc = [&](size_t bytes) -> void* {
        void* p = ws + off;
        off = (off + bytes + 255) & ~(size_t)255;
        return p;
    };
    float*     feat  = (float*)    alloc((size_t)BN * 64 * 4);
    _Float16*  featH = (_Float16*) alloc((size_t)BN * 64 * 2);
    float*     sq    = (float*)    alloc((size_t)BN * 4);
    int*       nbr   = (int*)      alloc((size_t)BN * NNBR * 4);
    float*     Wh    = (float*)    alloc((size_t)BN * 64 * 4);
    float*     s1    = (float*)    alloc((size_t)BN * 4);
    float*     s2    = (float*)    alloc((size_t)BN * 4);
    float*     hcat  = (float*)    alloc((size_t)BN * 128 * 4);
    _Float16*  hcatH = (_Float16*) alloc((size_t)BN * 128 * 2);
    _Float16*  gH    = (_Float16*) alloc((size_t)BN * 64 * 2);
    (void)ws_size; (void)in_sizes; (void)n_in; (void)out_size;

    const size_t conv_lds  = 25344 + 12672 + (size_t)(64 * 576) * 2;  // 111744 B
    const size_t tconv_lds = (size_t)(3 * 18 * 64) * 2 + (size_t)(64 * 576) * 2; // 80640 B

    // 1) stride-2 conv (async-LDS halo + one-shot f16 convert + 18 unrolled WMMAs)
    k_conv_s2_wmma<<<BATCH * 144, 128, conv_lds, stream>>>(x, conv_w, conv_b, feat, featH);
    // 2) squared norms
    k_rowsq<<<(BN + 255) / 256, 256, 0, stream>>>(feat, sq);
    // 3) WMMA Gram stripes in LDS + stable 7-NN selection (147456 B dynamic LDS)
    k_knn7_wmma<<<BATCH * 144, 256, 16 * NNODE * sizeof(float), stream>>>(featH, sq, nbr);

    // 4) GAT head 1 -> hcat[:, 0:64]
    k_gemm_wmma<64><<<BN / 16, 128, 0, stream>>>(featH, W1, Wh);
    k_attn_scores<<<(BN + 255) / 256, 256, 0, stream>>>(Wh, a1, s1, s2);
    k_gat_aggregate<<<(BN * 64) / 256, 256, 0, stream>>>(Wh, s1, s2, nbr, hcat, hcatH, 128, 0, 1, 0);
    // 5) GAT head 2 -> hcat[:, 64:128]
    k_gemm_wmma<64><<<BN / 16, 128, 0, stream>>>(featH, W2, Wh);
    k_attn_scores<<<(BN + 255) / 256, 256, 0, stream>>>(Wh, a2, s1, s2);
    k_gat_aggregate<<<(BN * 64) / 256, 256, 0, stream>>>(Wh, s1, s2, nbr, hcat, hcatH, 128, 64, 1, 0);
    // 6) output attention layer (K=128) -> gH in f16 [B][C][N], ELU applied
    k_gemm_wmma<128><<<BN / 16, 128, 0, stream>>>(hcatH, Wout, Wh);
    k_attn_scores<<<(BN + 255) / 256, 256, 0, stream>>>(Wh, aout, s1, s2);
    k_gat_aggregate<<<(BN * 64) / 256, 256, 0, stream>>>(Wh, s1, s2, nbr, nullptr, gH, 0, 0, 1, 1);
    // 7) transpose conv as WMMA implicit GEMM over the dilated grid
    k_tconv_wmma<<<BATCH * 96 * 6, 128, tconv_lds, stream>>>(gH, tw, tb, (float*)d_out);
}
